// ChebGraphConv_SC_FC_age_ordinal_scanner_10110353015360
// MI455X (gfx1250) — matrix operand order repaired and checked by reading, not compiled, not run on hardware
//
#include <hip/hip_runtime.h>

// ---------------------------------------------------------------------------
// MI455X (gfx1250, wave32) ChebGraphConv forward.
// All GEMMs -> v_wmma_f32_16x16x32_bf16 (fp32 accum). Every WMMA fragment is
// two 16-byte b128 loads (K contiguous per lane) via dual row-major /
// transposed bf16 copies. One wave = 32x32 output tile (2x2 WMMA blocking);
// the 8 waves of a block are arranged 2x4 (or 4x2) so the block's 64x128
// (128x64) macro-tile dedups both A and B through the WGP L0 cache:
// ~44 FLOP per byte of L2 traffic for the dominant 512^3 batched GEMMs.
// ---------------------------------------------------------------------------

typedef __attribute__((ext_vector_type(16))) __bf16    v16bf;
typedef __attribute__((ext_vector_type(8)))  float     v8f;
typedef __attribute__((ext_vector_type(4)))  unsigned  uint4v;

union FragU {
  v16bf v;
  uint4v q[2];
  unsigned u[8];
  unsigned short s[16];
};

__device__ __forceinline__ unsigned short f2bf(float f) {
  unsigned u = __builtin_bit_cast(unsigned, f);
  unsigned r = u + 0x7FFFu + ((u >> 16) & 1u);   // round-to-nearest-even
  return (unsigned short)(r >> 16);
}
__device__ __forceinline__ float bf2f(unsigned short h) {
  unsigned u = ((unsigned)h) << 16;
  return __builtin_bit_cast(float, u);
}

// ---- WMMA fragment loaders (CDNA5 ISA §7.12.2), all-b128 -------
// A fragment 16x32: lane l: row=l&15, half=l>>4. K pairs: VGPR0-3 K=8h+{0..7},
// VGPR4-7 K=16+8h+{0..7} -> two contiguous 16B loads at +kb and +kb+16.
__device__ __forceinline__ v16bf load_ak(const unsigned short* p0, int ld) {
  const int lane = threadIdx.x & 31;
  const unsigned short* p = p0 + (size_t)(lane & 15) * ld + ((lane >> 4) << 3);
  FragU f;
  f.q[0] = *reinterpret_cast<const uint4v*>(p);
  f.q[1] = *reinterpret_cast<const uint4v*>(p + 16);
  return f.v;
}
// B fragment 32x16 loaded from "B^T row-major" storage (column c's K values
// contiguous): lane l: col=l&15, half=l>>4: K=8h+{0..7}, 16+8h+{0..7}
// -> two contiguous 16B loads at +kb and +kb+16.
__device__ __forceinline__ v16bf load_bk(const unsigned short* p0, int ld) {
  const int lane = threadIdx.x & 31;
  const unsigned short* p = p0 + (size_t)(lane & 15) * ld + ((lane >> 4) << 3);
  FragU f;
  f.q[0] = *reinterpret_cast<const uint4v*>(p);
  f.q[1] = *reinterpret_cast<const uint4v*>(p + 16);
  return f.v;
}

__device__ __forceinline__ v8f wmma_bf16(v16bf a, v16bf b, v8f c) {
  return __builtin_amdgcn_wmma_f32_16x16x32_bf16(false, a, false, b,
                                                 (short)0, c, false, false);
}

// ---------------------------------------------------------------------------
// Kernels
// ---------------------------------------------------------------------------

__global__ void k_f32_to_bf16(const float* __restrict__ src,
                              unsigned short* __restrict__ dst, int n) {
  int i = blockIdx.x * 256 + threadIdx.x;
  const int stride = gridDim.x * 256;
  for (; i < n; i += stride) dst[i] = f2bf(src[i]);
}

// Tiled transpose with bf16 convert: src f32 [batch,R,C] -> dst bf16 [batch,C,R]
__global__ __launch_bounds__(256) void k_transpose_bf16(
    const float* __restrict__ src, unsigned short* __restrict__ dst,
    int R, int C) {
  __shared__ unsigned short tile[32][33];
  const int tx = threadIdx.x;            // 0..31
  const int ty = threadIdx.y;            // 0..7
  const size_t zb = (size_t)blockIdx.z * R * C;
  const int c0 = blockIdx.x * 32, r0 = blockIdx.y * 32;
  for (int j = ty; j < 32; j += 8)
    tile[j][tx] = f2bf(src[zb + (size_t)(r0 + j) * C + c0 + tx]);
  __syncthreads();
  for (int j = ty; j < 32; j += 8)
    dst[zb + (size_t)(c0 + j) * R + r0 + tx] = tile[tx][j];
}

// Epilogue helper: write one 16x16 sub-tile to transposed (contiguous b128)
// and row-major (8 strided u16) layouts, with optional prev2 subtract.
__device__ __forceinline__ void epi_store(
    v8f acc, float scale, int sub,
    const unsigned short* __restrict__ Tprev2T,
    unsigned short* __restrict__ ToutT,
    unsigned short* __restrict__ ToutR,
    size_t trb, size_t rwb, int Nc, int m0, int n0) {
  const int lane = threadIdx.x & 31;
  const int n  = lane & 15;
  const int mb = (lane >> 4) << 3;
  const int ng = n0 + n;
  const int mg = m0 + mb;
  const size_t tbase = trb + (size_t)ng * 512 + mg;   // 8 contiguous elements
  float pv[8] = {0.f, 0.f, 0.f, 0.f, 0.f, 0.f, 0.f, 0.f};
  if (sub) {
    uint4v pq = *reinterpret_cast<const uint4v*>(Tprev2T + tbase);
    const unsigned short* ps = reinterpret_cast<const unsigned short*>(&pq);
#pragma unroll
    for (int i = 0; i < 8; ++i) pv[i] = bf2f(ps[i]);
  }
  FragU o;
#pragma unroll
  for (int i = 0; i < 8; ++i) o.s[i] = f2bf(scale * acc[i] - pv[i]);
  *reinterpret_cast<uint4v*>(ToutT + tbase) = o.q[0];
#pragma unroll
  for (int i = 0; i < 8; ++i)
    ToutR[rwb + (size_t)(mg + i) * Nc + ng] = o.s[i];
}

// T_out = scale*(A @ T_prev) - (sub ? T_prev2 : 0).
// A: [B,512,512] bf16 row-major. T_prev/T_prev2/T_out transposed [B,Nc,512];
// T_out also written row-major [B,512,Nc]. One wave = 32x32 output tile;
// waves arranged 2x4 (Nc>=128) or 4x2 (Nc==64) for L0 operand sharing.
__global__ __launch_bounds__(256) void k_cheb_gemm(
    const unsigned short* __restrict__ Abf,
    const unsigned short* __restrict__ TprevT,
    const unsigned short* __restrict__ Tprev2T,
    unsigned short* __restrict__ ToutT,
    unsigned short* __restrict__ ToutR,
    int Nc, float scale, int sub) {
  const int wave = threadIdx.x >> 5;
  const int b = blockIdx.y;
  int m0, n0;
  if (Nc >= 128) {                     // block tile 64(M) x 128(N), waves 2x4
    const int bTilesN = Nc >> 7;
    const int bm = blockIdx.x / bTilesN;
    const int bn = blockIdx.x - bm * bTilesN;
    m0 = bm * 64 + (wave >> 2) * 32;
    n0 = bn * 128 + (wave & 3) * 32;
  } else {                             // block tile 128(M) x 64(N), waves 4x2
    m0 = blockIdx.x * 128 + (wave >> 1) * 32;
    n0 = (wave & 1) * 32;
  }

  const unsigned short* Ab = Abf    + (size_t)b * (512 * 512) + (size_t)m0 * 512;
  const unsigned short* Bb = TprevT + (size_t)b * Nc * 512    + (size_t)n0 * 512;

  v8f acc00 = {0.f,0.f,0.f,0.f,0.f,0.f,0.f,0.f};
  v8f acc01 = acc00, acc10 = acc00, acc11 = acc00;
  for (int kk = 0; kk < 512; kk += 32) {
    v16bf a0 = load_ak(Ab + kk, 512);
    v16bf a1 = load_ak(Ab + (size_t)16 * 512 + kk, 512);
    v16bf b0 = load_bk(Bb + kk, 512);
    v16bf b1 = load_bk(Bb + (size_t)16 * 512 + kk, 512);
    acc00 = wmma_bf16(a0, b0, acc00);
    acc01 = wmma_bf16(a0, b1, acc01);
    acc10 = wmma_bf16(a1, b0, acc10);
    acc11 = wmma_bf16(a1, b1, acc11);
  }
  const size_t trb = (size_t)b * Nc * 512;
  const size_t rwb = (size_t)b * 512 * Nc;
  epi_store(acc00, scale, sub, Tprev2T, ToutT, ToutR, trb, rwb, Nc, m0,      n0);
  epi_store(acc01, scale, sub, Tprev2T, ToutT, ToutR, trb, rwb, Nc, m0,      n0 + 16);
  epi_store(acc10, scale, sub, Tprev2T, ToutT, ToutR, trb, rwb, Nc, m0 + 16, n0);
  epi_store(acc11, scale, sub, Tprev2T, ToutT, ToutR, trb, rwb, Nc, m0 + 16, n0 + 16);
}

// OUT[B,512,64] (+)= T_row[B,512,Kd] @ W ; W given transposed WT[64,Kd] bf16.
// Block tile 128(M) x 64(N), waves 4x2.
__global__ __launch_bounds__(256) void k_weight_accum(
    const unsigned short* __restrict__ Trow,
    const unsigned short* __restrict__ WT,
    const float* __restrict__ bias,
    float* __restrict__ OUT,
    int Kd, int init) {
  const int wave = threadIdx.x >> 5;
  const int b  = blockIdx.y;
  const int m0 = blockIdx.x * 128 + (wave >> 1) * 32;
  const int n0 = (wave & 1) * 32;

  const unsigned short* Ab = Trow + (size_t)b * 512 * Kd + (size_t)m0 * Kd;
  const unsigned short* Bb = WT + (size_t)n0 * Kd;

  v8f acc00 = {0.f,0.f,0.f,0.f,0.f,0.f,0.f,0.f};
  v8f acc01 = acc00, acc10 = acc00, acc11 = acc00;
  for (int kk = 0; kk < Kd; kk += 32) {
    v16bf a0 = load_ak(Ab + kk, Kd);
    v16bf a1 = load_ak(Ab + (size_t)16 * Kd + kk, Kd);
    v16bf b0 = load_bk(Bb + kk, Kd);
    v16bf b1 = load_bk(Bb + (size_t)16 * Kd + kk, Kd);
    acc00 = wmma_bf16(a0, b0, acc00);
    acc01 = wmma_bf16(a0, b1, acc01);
    acc10 = wmma_bf16(a1, b0, acc10);
    acc11 = wmma_bf16(a1, b1, acc11);
  }
  const int lane = threadIdx.x & 31;
  const int n  = lane & 15;
  const int mb = (lane >> 4) << 3;
  const size_t ob = (size_t)b * 512 * 64;
  v8f accs[4] = {acc00, acc01, acc10, acc11};
#pragma unroll
  for (int t = 0; t < 4; ++t) {
    const int mg = m0 + (t >> 1) * 16 + mb;
    const int ng = n0 + (t & 1) * 16 + n;
#pragma unroll
    for (int i = 0; i < 8; ++i) {
      const size_t idx = ob + (size_t)(mg + i) * 64 + ng;
      const float base = init ? bias[ng] : OUT[idx];
      OUT[idx] = base + accs[t][i];
    }
  }
}

// AE = relu(X0 @ X0^T) with zeroed diagonal. relu(G) is exactly symmetric so
// the reference's 0.5*(AE+AE^T) is the identity and is skipped.
// Block tile 64(M) x 128(N), waves 2x4.
__global__ __launch_bounds__(256) void k_outer_ae(
    const unsigned short* __restrict__ X0bf, float* __restrict__ AE) {
  const int wave = threadIdx.x >> 5;
  const int b  = blockIdx.y;
  const int bm = blockIdx.x >> 2;            // 512/64 = 8
  const int bn = blockIdx.x & 3;             // 512/128 = 4
  const int m0 = bm * 64 + (wave >> 2) * 32;
  const int n0 = bn * 128 + (wave & 3) * 32;

  const unsigned short* Xb = X0bf + (size_t)b * 512 * 64;
  const unsigned short* Ar = Xb + (size_t)m0 * 64;
  const unsigned short* Br = Xb + (size_t)n0 * 64;

  v8f acc00 = {0.f,0.f,0.f,0.f,0.f,0.f,0.f,0.f};
  v8f acc01 = acc00, acc10 = acc00, acc11 = acc00;
  for (int kk = 0; kk < 64; kk += 32) {
    v16bf a0 = load_ak(Ar + kk, 64);
    v16bf a1 = load_ak(Ar + (size_t)16 * 64 + kk, 64);
    v16bf b0 = load_bk(Br + kk, 64);
    v16bf b1 = load_bk(Br + (size_t)16 * 64 + kk, 64);
    acc00 = wmma_bf16(a0, b0, acc00);
    acc01 = wmma_bf16(a0, b1, acc01);
    acc10 = wmma_bf16(a1, b0, acc10);
    acc11 = wmma_bf16(a1, b1, acc11);
  }
  const int lane = threadIdx.x & 31;
  const int n  = lane & 15;
  const int mb = (lane >> 4) << 3;
  v8f accs[4] = {acc00, acc01, acc10, acc11};
#pragma unroll
  for (int t = 0; t < 4; ++t) {
    const int mg0 = m0 + (t >> 1) * 16 + mb;
    const int ng  = n0 + (t & 1) * 16 + n;
#pragma unroll
    for (int i = 0; i < 8; ++i) {
      const int mg = mg0 + i;
      float v = fmaxf(accs[t][i], 0.f);
      if (mg == ng) v = 0.f;
      AE[(size_t)b * 512 * 512 + (size_t)mg * 512 + ng] = v;
    }
  }
}

// LayerNorm over [N,D]=[512,64] per batch + ReLU -> fp32 and row-major bf16.
__global__ __launch_bounds__(256) void k_ln_relu(
    const float* __restrict__ IN, const float* __restrict__ g,
    const float* __restrict__ bt, float* __restrict__ Hf,
    unsigned short* __restrict__ Hrow) {
  const int b = blockIdx.x;
  const int t = threadIdx.x;
  const float* x = IN + (size_t)b * 32768;
  float s = 0.f, s2 = 0.f;
  for (int i = t; i < 32768; i += 256) {
    float v = x[i];
    s += v; s2 += v * v;
  }
  __shared__ float sh[256], sh2[256];
  sh[t] = s; sh2[t] = s2;
  __syncthreads();
  for (int o = 128; o > 0; o >>= 1) {
    if (t < o) { sh[t] += sh[t + o]; sh2[t] += sh2[t + o]; }
    __syncthreads();
  }
  const float mean = sh[0] * (1.f / 32768.f);
  const float var  = sh2[0] * (1.f / 32768.f) - mean * mean;
  const float rstd = rsqrtf(var + 1e-5f);
  for (int i = t; i < 32768; i += 256) {
    float v = (x[i] - mean) * rstd * g[i] + bt[i];
    v = fmaxf(v, 0.f);
    Hf[(size_t)b * 32768 + i]   = v;
    Hrow[(size_t)b * 32768 + i] = f2bf(v);
  }
}

__global__ void k_pool(const float* __restrict__ Hf, float* __restrict__ Z,
                       int off) {
  const int b = blockIdx.x;
  const int f = threadIdx.x;   // 64 threads
  const float* x = Hf + (size_t)b * 32768 + f;
  float s = 0.f, mx = -3.402823466e38f;
  for (int nn = 0; nn < 512; ++nn) {
    float v = x[(size_t)nn * 64];
    s += v; mx = fmaxf(mx, v);
  }
  Z[b * 256 + off + f]      = s * (1.f / 512.f);
  Z[b * 256 + off + 64 + f] = mx;
}

__device__ __forceinline__ float lrelu(float x) {
  return x >= 0.f ? x : 0.2f * x;
}

__global__ void k_heads(const float* __restrict__ Z,
                        const float* __restrict__ p1w, const float* __restrict__ p1b,
                        const float* __restrict__ p2w, const float* __restrict__ p2b,
                        const float* __restrict__ p3w, const float* __restrict__ p3b,
                        const float* __restrict__ s1w, const float* __restrict__ s1b,
                        const float* __restrict__ s2w, const float* __restrict__ s2b,
                        const float* __restrict__ p4w, const float* __restrict__ p4b,
                        float* __restrict__ out_class,
                        float* __restrict__ scanner) {
  const int b = blockIdx.x;
  const int f = threadIdx.x;   // 64 threads
  __shared__ float z[256], t0[64], t1[64], lg[2];
  for (int i = f; i < 256; i += 64) z[i] = Z[b * 256 + i];
  __syncthreads();

  float a = p1b[f];
  for (int i = 0; i < 256; ++i) a += z[i] * p1w[i * 64 + f];
  t0[f] = lrelu(a);
  __syncthreads();
  a = p2b[f];
  for (int i = 0; i < 64; ++i) a += t0[i] * p2w[i * 64 + f];
  t1[f] = lrelu(a);
  __syncthreads();
  if (f < 4) {
    float o = p3b[f];
    for (int i = 0; i < 64; ++i) o += t1[i] * p3w[i * 4 + f];
    out_class[b * 4 + f] = 1.f / (1.f + expf(-o));
  }
  float a3 = s1b[f];
  for (int i = 0; i < 256; ++i) a3 += z[i] * s1w[i * 64 + f];
  __syncthreads();
  t0[f] = lrelu(a3);
  __syncthreads();
  a = s2b[f];
  for (int i = 0; i < 64; ++i) a += t0[i] * s2w[i * 64 + f];
  t1[f] = lrelu(a);
  __syncthreads();
  if (f < 2) {
    float o = p4b[f];
    for (int i = 0; i < 64; ++i) o += t1[i] * p4w[i * 2 + f];
    lg[f] = o;
  }
  __syncthreads();
  if (f < 2) {
    const float m = fmaxf(lg[0], lg[1]);
    const float lse = m + logf(expf(lg[0] - m) + expf(lg[1] - m));
    scanner[b * 2 + f] = lg[f] - lse;
  }
}

// ---------------------------------------------------------------------------
// Host orchestration
// ---------------------------------------------------------------------------

extern "C" void kernel_launch(void* const* d_in, const int* in_sizes, int n_in,
                              void* d_out, int out_size, void* d_ws,
                              size_t ws_size, hipStream_t stream) {
  (void)in_sizes; (void)n_in; (void)out_size; (void)ws_size;
  constexpr int B = 64, N = 512, D = 64;

  const float* A   = (const float*)d_in[0];
  const float* X   = (const float*)d_in[1];
  const float* W1  = (const float*)d_in[2];
  const float* b1  = (const float*)d_in[3];
  const float* W2  = (const float*)d_in[4];
  const float* b2  = (const float*)d_in[5];
  const float* W3  = (const float*)d_in[6];
  const float* b3  = (const float*)d_in[7];
  const float* g1  = (const float*)d_in[8];
  const float* bt1 = (const float*)d_in[9];
  const float* g2  = (const float*)d_in[10];
  const float* bt2 = (const float*)d_in[11];
  const float* p1w = (const float*)d_in[12];
  const float* p1b = (const float*)d_in[13];
  const float* p2w = (const float*)d_in[14];
  const float* p2b = (const float*)d_in[15];
  const float* p3w = (const float*)d_in[16];
  const float* p3b = (const float*)d_in[17];
  const float* s1w = (const float*)d_in[18];
  const float* s1b = (const float*)d_in[19];
  const float* s2w = (const float*)d_in[20];
  const float* s2b = (const float*)d_in[21];
  const float* p4w = (const float*)d_in[22];
  const float* p4b = (const float*)d_in[23];

  char* ws = (char*)d_ws;
  size_t off = 0;
  auto carve = [&](size_t bytes) -> void* {
    void* p = ws + off;
    off += (bytes + 255) & ~(size_t)255;
    return p;
  };
  const size_t BNN2 = (size_t)B * N * N * 2;
  unsigned short* Abf = (unsigned short*)carve(BNN2);
  unsigned short* TR[3], *TT[3];             // rotating row/transposed T pairs
  for (int i = 0; i < 3; ++i) {
    TR[i] = (unsigned short*)carve(BNN2);
    TT[i] = (unsigned short*)carve(BNN2);
  }
  float*          OUT   = (float*)carve((size_t)B * N * D * 4);
  float*          Hf    = (float*)carve((size_t)B * N * D * 4);
  unsigned short* Hrow  = (unsigned short*)carve((size_t)B * N * D * 2);
  unsigned short* Htr   = (unsigned short*)carve((size_t)B * N * D * 2);
  unsigned short* X0row = (unsigned short*)carve((size_t)B * N * D * 2);
  float*          Zb    = (float*)carve((size_t)B * 256 * 4);
  unsigned short* W1T   = (unsigned short*)carve((size_t)6 * N * D * 2);
  unsigned short* W2T   = (unsigned short*)carve((size_t)6 * D * D * 2);
  unsigned short* W3T   = (unsigned short*)carve((size_t)6 * D * D * 2);

  auto cvt = [&](const float* s, unsigned short* d, int n) {
    int blocks = (n + 255) / 256;
    if (blocks > 2048) blocks = 2048;
    k_f32_to_bf16<<<dim3(blocks), dim3(256), 0, stream>>>(s, d, n);
  };
  auto tr = [&](const float* s, unsigned short* d, int R, int C, int batch) {
    k_transpose_bf16<<<dim3(C / 32, R / 32, batch), dim3(32, 8), 0, stream>>>(s, d, R, C);
  };

  cvt(A, Abf, B * N * N);
  cvt(X, TR[0], B * N * N);                 // X row-major bf16
  tr(X, TT[0], N, N, B);                    // X transposed bf16
  tr(W1, W1T, N, D, 6);                     // W^T [6,64,Kd]
  tr(W2, W2T, D, D, 6);
  tr(W3, W3T, D, D, 6);

  // One Chebyshev layer. T0 given as (T0r,T0t); pair slots p0/p1/p2 rotate
  // (p0 may alias T0). Wb = W^T [6,64,Kd] bf16, bias [64].
  auto cheb = [&](const unsigned short* T0r, const unsigned short* T0t,
                  unsigned short* p0r, unsigned short* p0t,
                  unsigned short* p1r, unsigned short* p1t,
                  unsigned short* p2r, unsigned short* p2t,
                  const unsigned short* Wb, const float* bias, int Nc, int Kd) {
    // Nc=512: (512/64)*(512/128)=32 blocks; Nc=64: 512/128=4 blocks.
    const dim3 ggrid(Nc >= 128 ? (512 / 64) * (Nc / 128) : 512 / 128, B);
    const dim3 wgrid(4, B);
    k_cheb_gemm<<<ggrid, 256, 0, stream>>>(Abf, T0t, (const unsigned short*)nullptr,
                                           p1t, p1r, Nc, 1.f, 0);
    k_weight_accum<<<wgrid, 256, 0, stream>>>(T0r, Wb, bias, OUT, Kd, 1);
    k_weight_accum<<<wgrid, 256, 0, stream>>>(p1r, Wb + (size_t)D * Kd,
                                              (const float*)nullptr, OUT, Kd, 0);
    const unsigned short* prev2t = T0t;
    const unsigned short* prevt  = p1t;
    unsigned short* curR[4] = {p2r, p0r, p1r, p2r};
    unsigned short* curT[4] = {p2t, p0t, p1t, p2t};
    for (int k = 2; k <= 5; ++k) {
      k_cheb_gemm<<<ggrid, 256, 0, stream>>>(Abf, prevt, prev2t, curT[k - 2],
                                             curR[k - 2], Nc, 2.f, 1);
      k_weight_accum<<<wgrid, 256, 0, stream>>>(curR[k - 2], Wb + (size_t)k * D * Kd,
                                                (const float*)nullptr, OUT, Kd, 0);
      prev2t = prevt;
      prevt  = curT[k - 2];
    }
  };

  // Layer 1: input X [B,512,512], W1 [6,512,64]
  cheb(TR[0], TT[0], TR[0], TT[0], TR[1], TT[1], TR[2], TT[2], W1T, b1, N, N);
  k_ln_relu<<<dim3(B), 256, 0, stream>>>(OUT, g1, bt1, Hf, Hrow);
  tr(Hf, Htr, N, D, B);
  k_pool<<<dim3(B), 64, 0, stream>>>(Hf, Zb, 0);

  // Layer 2: input h [B,512,64], W2 [6,64,64]
  cheb(Hrow, Htr, TR[0], TT[0], TR[1], TT[1], TR[2], TT[2], W2T, b2, D, D);
  k_ln_relu<<<dim3(B), 256, 0, stream>>>(OUT, g2, bt2, Hf, Hrow);
  tr(Hf, Htr, N, D, B);
  k_pool<<<dim3(B), 64, 0, stream>>>(Hf, Zb, 128);

  // Layer 3: end_gcn (no LN / relu)
  cheb(Hrow, Htr, TR[0], TT[0], TR[1], TT[1], TR[2], TT[2], W3T, b3, D, D);
  cvt(OUT, X0row, B * N * D);

  float* AE = (float*)d_out;
  k_outer_ae<<<dim3(32, B), 256, 0, stream>>>(X0row, AE);

  float* out_class = AE + (size_t)B * N * N;
  float* scanner   = out_class + (size_t)B * 4;
  k_heads<<<dim3(B), 64, 0, stream>>>(Zb, p1w, p1b, p2w, p2b, p3w, p3b,
                                      s1w, s1b, s2w, s2b, p4w, p4b,
                                      out_class, scanner);
}
